// Decoder_28527172780196
// MI455X (gfx1250) — compile-verified, matrix operand order checked by reading, not run on hardware
//
#include <hip/hip_runtime.h>
#include <hip/hip_bf16.h>

// ---------------------------------------------------------------------------
// GRU decoder on gfx1250 (MI455X), bf16 WMMA pipeline:
//   1) convert W_ih / W_hh / W_out fp32 -> bf16 (one-time; W_out bf16 = 65MB,
//      fits in the 192MB L2)
//   2) gather embeddings -> bf16
//   3) xW = x @ W_ih^T + b_ih      (LDS-tiled WMMA GEMM, 4096x3072, K=512)
//   4) 64 sequential fused GRU steps (one wave owns a 16x16 block: 3 WMMA
//      tiles + gate math, double-buffered h)
//   5) logits = hs @ W_out^T + b_out (LDS-tiled WMMA GEMM, 4096x32000,
//      K=1024; 268 GFLOP, the dominant cost)
//
// GEMM tiling: 256-thread block = 8 waves covers 128M x 128N. K is staged
// 32-deep into LDS (A-tile 8KB + B-tile 8KB, double buffered = 40KB with
// padding). Per stage each wave does 12 ds_load_b128 feeding 8
// v_wmma_f32_16x16x32_bf16; each thread does 4 global b128 loads. This cuts
// L2 traffic for the logits GEMM from ~16.6GB (direct) to ~4.1GB.
// ---------------------------------------------------------------------------

typedef __attribute__((ext_vector_type(16))) __bf16 v16bf;
typedef __attribute__((ext_vector_type(8)))  float  v8f;

static constexpr int GV = 32000, GE = 512, GH = 1024, GB = 64, GT = 64;

// LDS tile geometry: 128 rows x 32 bf16 per stage, row padded to 80 bytes so
// (row*20) mod 64 banks is a perfect permutation -> conflict-free ds_load.
static constexpr int LROW  = 80;               // bytes per LDS tile row
static constexpr int LTILE = 128 * LROW;       // 10240 B per A or B tile
static constexpr int LBUF  = 2 * LTILE;        // A + B for one stage

static __device__ __forceinline__ v8f wmma_bf16(v16bf a, v16bf b, v8f c) {
  return __builtin_amdgcn_wmma_f32_16x16x32_bf16(false, a, false, b, (short)0, c, false, false);
}

// A/B fragment for v_wmma_*_16x16x32_bf16 from global row-major memory.
// Lane l holds row (l&15), K chunks [koff..koff+7] and [koff+16..koff+23],
// koff = k + 8*(l>>4). Caller pre-adds row*K + 8*(l>>4) into `base`.
static __device__ __forceinline__ v16bf gmem_frag(const __bf16* __restrict__ base, int k) {
  union { v16bf v; uint4 q[2]; } f;
  f.q[0] = *reinterpret_cast<const uint4*>(base + k);
  f.q[1] = *reinterpret_cast<const uint4*>(base + k + 16);
  return f.v;
}

// Same fragment out of an LDS stage tile: chunk0 at row*80 + 16*half,
// chunk1 32 bytes later. Caller passes p = tile + row*80 + 16*half.
static __device__ __forceinline__ v16bf lds_frag(const unsigned char* p) {
  union { v16bf v; uint4 q[2]; } f;
  f.q[0] = *reinterpret_cast<const uint4*>(p);
  f.q[1] = *reinterpret_cast<const uint4*>(p + 32);
  return f.v;
}

// ---------------------------------------------------------------------------
// fp32 -> bf16 bulk convert (grid-stride)
__global__ void f32_to_bf16_kernel(const float* __restrict__ s, __bf16* __restrict__ d, long n) {
  long i = (long)blockIdx.x * blockDim.x + threadIdx.x;
  long stride = (long)gridDim.x * blockDim.x;
  for (; i < n; i += stride) d[i] = (__bf16)s[i];
}

// Gather embedding rows for shifted targets (SOS=1 at t=0), fp32 -> bf16
__global__ void gather_embed_kernel(const int* __restrict__ targets,
                                    const float* __restrict__ emb,
                                    __bf16* __restrict__ x) {
  const int tb = blockIdx.x;            // 0 .. T*B-1
  const int t = tb / GB, b = tb % GB;
  const int tok = (t == 0) ? 1 : targets[(t - 1) * GB + b];
  const float* __restrict__ row = emb + (size_t)tok * GE;
  __bf16* __restrict__ out = x + (size_t)tb * GE;
  for (int e = threadIdx.x; e < GE; e += blockDim.x) out[e] = (__bf16)row[e];
}

// h0 = encode_vec[0]  (fp32 copy + bf16 shadow for the first WMMA step)
__global__ void init_h0_kernel(const float* __restrict__ enc,
                               float* __restrict__ h_f, __bf16* __restrict__ h_b, int n) {
  int i = blockIdx.x * blockDim.x + threadIdx.x;
  int stride = gridDim.x * blockDim.x;
  for (; i < n; i += stride) { float v = enc[i]; h_f[i] = v; h_b[i] = (__bf16)v; }
}

// ---------------------------------------------------------------------------
// C[M,N] = A[M,K] @ B[N,K]^T + bias[N]   (A,B bf16 row-major, C fp32)
// Block: 256 threads / 8 waves, 128M x 128N tile. Waves arranged 2(M) x 4(N);
// each wave owns 4 M-tiles x 2 N-tiles = 8 f32 accumulators (64 VGPRs).
// K staged 32-deep in double-buffered LDS; one barrier per stage.
// M, N, K must be multiples of 128, 128, 32 (true for all call sites), so
// there is no tail code and EXEC is all-ones around every WMMA.
__global__ __launch_bounds__(256) void gemm_lds_bf16_kernel(
    const __bf16* __restrict__ A, const __bf16* __restrict__ Bm,
    const float* __restrict__ bias, float* __restrict__ C,
    int N, int K) {
  __shared__ __align__(16) unsigned char smem[2 * LBUF];   // 40 KB

  const int lane = threadIdx.x & 31;
  const int wave = threadIdx.x >> 5;
  const int half = lane >> 4;                 // which K-half this lane holds
  const int lr   = lane & 15;
  const int mw   = wave & 1;                  // 0/1 -> 64-row group
  const int nw   = wave >> 1;                 // 0..3 -> 32-col group
  const int m0 = blockIdx.y << 7;
  const int n0 = blockIdx.x << 7;

  // --- staging assignment: thread -> one 32B piece of A and of B per stage
  const int srow = threadIdx.x >> 1;          // 0..127
  const int sc   = threadIdx.x & 1;           // 0/1 -> element halves [16c,16c+16)
  const __bf16* __restrict__ gA = A  + (size_t)(m0 + srow) * K + 16 * sc;
  const __bf16* __restrict__ gB = Bm + (size_t)(n0 + srow) * K + 16 * sc;
  const int soff = srow * LROW + 32 * sc;

  // --- compute-side LDS offsets (fragment base per wave/lane)
  const int aoff = (mw * 64 + lr) * LROW + half * 16;
  const int boff = (nw * 32 + lr) * LROW + half * 16;

  v8f acc[4][2];
#pragma unroll
  for (int mt = 0; mt < 4; ++mt)
#pragma unroll
    for (int nt = 0; nt < 2; ++nt) acc[mt][nt] = v8f{};

  const int nk = K >> 5;                      // stages of 32

  // Prologue: stage 0 into buffer 0.
  {
    uint4 a0 = *reinterpret_cast<const uint4*>(gA);
    uint4 a1 = *reinterpret_cast<const uint4*>(gA + 8);
    uint4 b0 = *reinterpret_cast<const uint4*>(gB);
    uint4 b1 = *reinterpret_cast<const uint4*>(gB + 8);
    unsigned char* qa = smem + soff;
    unsigned char* qb = smem + LTILE + soff;
    *reinterpret_cast<uint4*>(qa)      = a0;
    *reinterpret_cast<uint4*>(qa + 16) = a1;
    *reinterpret_cast<uint4*>(qb)      = b0;
    *reinterpret_cast<uint4*>(qb + 16) = b1;
  }
  __syncthreads();

  for (int ks = 0; ks < nk; ++ks) {
    const int buf  = ks & 1;
    const bool more = (ks + 1) < nk;

    // Issue next stage's global loads first so they overlap the WMMA chain.
    uint4 a0, a1, b0, b1;
    if (more) {
      const int kn = (ks + 1) << 5;
      a0 = *reinterpret_cast<const uint4*>(gA + kn);
      a1 = *reinterpret_cast<const uint4*>(gA + kn + 8);
      b0 = *reinterpret_cast<const uint4*>(gB + kn);
      b1 = *reinterpret_cast<const uint4*>(gB + kn + 8);
      __builtin_prefetch(gB + kn + 32, 0, 1);     // stream-ahead hint
    }

    // Compute current stage from LDS: 12 ds_load_b128 -> 8 WMMA.
    const unsigned char* pa = smem + buf * LBUF;
    const unsigned char* pb = pa + LTILE;
    v16bf af[4], bfg[2];
#pragma unroll
    for (int mt = 0; mt < 4; ++mt) af[mt]  = lds_frag(pa + aoff + mt * 16 * LROW);
#pragma unroll
    for (int nt = 0; nt < 2; ++nt) bfg[nt] = lds_frag(pb + boff + nt * 16 * LROW);
#pragma unroll
    for (int mt = 0; mt < 4; ++mt)
#pragma unroll
      for (int nt = 0; nt < 2; ++nt)
        acc[mt][nt] = wmma_bf16(af[mt], bfg[nt], acc[mt][nt]);

    // Park next stage into the other buffer, then one barrier per stage.
    if (more) {
      unsigned char* qa = smem + (buf ^ 1) * LBUF + soff;
      unsigned char* qb = qa + LTILE;
      *reinterpret_cast<uint4*>(qa)      = a0;
      *reinterpret_cast<uint4*>(qa + 16) = a1;
      *reinterpret_cast<uint4*>(qb)      = b0;
      *reinterpret_cast<uint4*>(qb + 16) = b1;
    }
    __syncthreads();
  }

  // Epilogue: lane -> (m = v + 8*half, n = lr); 16 consecutive lanes store
  // 16 consecutive floats -> 64B coalesced segments.
#pragma unroll
  for (int nt = 0; nt < 2; ++nt) {
    const int n = n0 + nw * 32 + nt * 16 + lr;
    const float bv = bias[n];
#pragma unroll
    for (int mt = 0; mt < 4; ++mt) {
      const int mb = m0 + mw * 64 + mt * 16 + (half << 3);
#pragma unroll
      for (int v = 0; v < 8; ++v) {
        C[(size_t)(mb + v) * N + n] = acc[mt][nt][v] + bv;
      }
    }
  }
}

// ---------------------------------------------------------------------------
// One GRU step, fully fused: wave owns a 16(batch) x 16(hidden-col) block,
// computes its r/z/n GEMM tiles (3 accumulators share one A fragment) and
// the gate nonlinearities, writing h_new (fp32 + bf16) and hs[t] (bf16).
// h is double-buffered across steps so no cross-block races within a step.
__global__ __launch_bounds__(32) void gru_step_kernel(
    const __bf16* __restrict__ h_bf, const __bf16* __restrict__ Whh,
    const float* __restrict__ b_hh, const float* __restrict__ xWt,
    const float* __restrict__ h_in, float* __restrict__ h_out,
    __bf16* __restrict__ h_out_bf, __bf16* __restrict__ hs_t) {
  const int lane = threadIdx.x & 31;
  const int half = lane >> 4;
  const int lr   = lane & 15;
  const int j0 = blockIdx.x << 4;       // hidden column block (0..1023)
  const int b0 = blockIdx.y << 4;       // batch row block     (0..63)

  const __bf16* __restrict__ arow = h_bf + (size_t)(b0 + lr) * GH + (half << 3);
  const __bf16* __restrict__ brow = Whh  + (size_t)(j0 + lr) * GH + (half << 3);
  const size_t zone = (size_t)GH * GH;  // r/z/n blocks of W_hh

  v8f accR = {}, accZ = {}, accN = {};
  for (int k = 0; k < GH; k += 32) {
    v16bf a  = gmem_frag(arow, k);
    v16bf br = gmem_frag(brow,            k);
    v16bf bz = gmem_frag(brow +     zone, k);
    v16bf bn = gmem_frag(brow + 2 * zone, k);
    accR = wmma_bf16(a, br, accR);
    accZ = wmma_bf16(a, bz, accZ);
    accN = wmma_bf16(a, bn, accN);
  }

  const int j = j0 + lr;
  const float bhr = b_hh[j], bhz = b_hh[GH + j], bhn = b_hh[2 * GH + j];
#pragma unroll
  for (int v = 0; v < 8; ++v) {
    const int b = b0 + (half << 3) + v;
    const float* __restrict__ xrow = xWt + (size_t)b * (3 * GH);
    const float xr = xrow[j], xz = xrow[GH + j], xn = xrow[2 * GH + j];
    const float hr = accR[v] + bhr;
    const float hz = accZ[v] + bhz;
    const float hn = accN[v] + bhn;
    const float r = 1.0f / (1.0f + __expf(-(xr + hr)));
    const float z = 1.0f / (1.0f + __expf(-(xz + hz)));
    const float nn = tanhf(xn + r * hn);
    const float hp = h_in[(size_t)b * GH + j];
    const float hnew = (1.0f - z) * nn + z * hp;
    h_out[(size_t)b * GH + j] = hnew;
    const __bf16 hb = (__bf16)hnew;
    h_out_bf[(size_t)b * GH + j] = hb;
    hs_t[(size_t)b * GH + j] = hb;
  }
}

// ---------------------------------------------------------------------------
extern "C" void kernel_launch(void* const* d_in, const int* in_sizes, int n_in,
                              void* d_out, int out_size, void* d_ws, size_t ws_size,
                              hipStream_t stream) {
  const float* enc  = (const float*)d_in[0];   // (1,B,H)
  const int*   tgt  = (const int*)d_in[1];     // (T,B)
  const float* emb  = (const float*)d_in[2];   // (V,E)
  const float* Wih  = (const float*)d_in[3];   // (3H,E)
  const float* Whh  = (const float*)d_in[4];   // (3H,H)
  const float* bih  = (const float*)d_in[5];   // (3H)
  const float* bhh  = (const float*)d_in[6];   // (3H)
  const float* Wout = (const float*)d_in[7];   // (V,H)
  const float* bout = (const float*)d_in[8];   // (V)
  float* logits = (float*)d_out;               // (T,B,V) fp32

  char* ws = (char*)d_ws;
  size_t off = 0;
  auto alloc = [&](size_t bytes) -> void* {
    off = (off + 255) & ~(size_t)255;
    void* p = ws + off; off += bytes; return p;
  };
  __bf16* Wih_b  = (__bf16*)alloc((size_t)3 * GH * GE * 2);
  __bf16* Whh_b  = (__bf16*)alloc((size_t)3 * GH * GH * 2);
  __bf16* Wout_b = (__bf16*)alloc((size_t)GV * GH * 2);
  __bf16* x_b    = (__bf16*)alloc((size_t)GT * GB * GE * 2);
  float*  xW     = (float*)alloc((size_t)GT * GB * 3 * GH * 4);
  float*  h_f0   = (float*)alloc((size_t)GB * GH * 4);
  float*  h_f1   = (float*)alloc((size_t)GB * GH * 4);
  __bf16* h_b0   = (__bf16*)alloc((size_t)GB * GH * 2);
  __bf16* h_b1   = (__bf16*)alloc((size_t)GB * GH * 2);
  __bf16* hs_b   = (__bf16*)alloc((size_t)GT * GB * GH * 2);
  float*  h_f[2] = {h_f0, h_f1};
  __bf16* h_b[2] = {h_b0, h_b1};

  // 1) weight conversion (one-time; ~37M elements)
  f32_to_bf16_kernel<<<1024, 256, 0, stream>>>(Wih,  Wih_b,  (long)3 * GH * GE);
  f32_to_bf16_kernel<<<1024, 256, 0, stream>>>(Whh,  Whh_b,  (long)3 * GH * GH);
  f32_to_bf16_kernel<<<4096, 256, 0, stream>>>(Wout, Wout_b, (long)GV * GH);

  // 2) embedding gather + h0
  gather_embed_kernel<<<GT * GB, 128, 0, stream>>>(tgt, emb, x_b);
  init_h0_kernel<<<256, 256, 0, stream>>>(enc, h_f[0], h_b[0], GB * GH);

  // 3) xW = x @ W_ih^T + b_ih : M=4096, N=3072, K=512 (128x128 blocks)
  gemm_lds_bf16_kernel<<<dim3(3 * GH / 128, GT * GB / 128), 256, 0, stream>>>(
      x_b, Wih_b, bih, xW, 3 * GH, GE);

  // 4) sequential GRU scan (stream order carries the dependency)
  for (int t = 0; t < GT; ++t) {
    const int cur = t & 1, nxt = (t + 1) & 1;
    gru_step_kernel<<<dim3(GH / 16, GB / 16), 32, 0, stream>>>(
        h_b[cur], Whh_b, bhh, xW + (size_t)t * GB * 3 * GH,
        h_f[cur], h_f[nxt], h_b[nxt], hs_b + (size_t)t * GB * GH);
  }

  // 5) logits = hs @ W_out^T + b_out : M=4096, N=32000, K=1024 (268 GFLOP)
  gemm_lds_bf16_kernel<<<dim3(GV / 128, GT * GB / 128), 256, 0, stream>>>(
      hs_b, Wout_b, bout, logits, GV, GH);
}
